// MHA_59158879535590
// MI455X (gfx1250) — matrix operand order repaired and checked by reading, not compiled, not run on hardware
//
#include <hip/hip_runtime.h>
#include <stdint.h>

#define B_ 2
#define T_ 2048
#define C_ 1024
#define H_ 16
#define D_ 64

typedef __attribute__((ext_vector_type(16))) __bf16 v16bf;
typedef __attribute__((ext_vector_type(8)))  float  v8f;
typedef int v4i_vs __attribute__((vector_size(16)));

union FragBF {
    v16bf v;
    uint4 q[2];
};

static __device__ __forceinline__ v8f wmma_bf16(v16bf a, v16bf b, v8f c) {
    // D = A(16x32 bf16) * B(32x16 bf16) + C(16x16 f32)
    return __builtin_amdgcn_wmma_f32_16x16x32_bf16(
        false, a, false, b, (short)0, c, false, false);
}

// ---------------------------------------------------------------------------
// Async global->LDS copy (CDNA5 GLOBAL_LOAD_ASYNC_TO_LDS_B128, ASYNCcnt).
// Builtin signature (from clang diagnostic): param0 = addrspace(1) v4i*,
// param1 = addrspace(3) v4i*, then imm offset, imm cpol.
// Guarded so a toolchain without the builtin still compiles (sync fallback).
// ---------------------------------------------------------------------------
#if defined(__has_builtin)
#  if __has_builtin(__builtin_amdgcn_global_load_async_to_lds_b128) && \
      __has_builtin(__builtin_amdgcn_s_wait_asynccnt)
#    define HAVE_ASYNC_LDS 1
#  endif
#endif

#ifdef HAVE_ASYNC_LDS
#define ASYNC_COPY_B128(g, l)                                                 \
    __builtin_amdgcn_global_load_async_to_lds_b128(                           \
        (__attribute__((address_space(1))) v4i_vs*)(uintptr_t)(g),            \
        (__attribute__((address_space(3))) v4i_vs*)(l), 0, 0)
#define ASYNC_WAIT(n) __builtin_amdgcn_s_wait_asynccnt(n)
#else
#define ASYNC_COPY_B128(g, l) (*(uint4*)(l) = *(const uint4*)(g))
#define ASYNC_WAIT(n) ((void)0)
#endif

// ---------------------------------------------------------------------------
// fp32 -> bf16 conversion
// ---------------------------------------------------------------------------
__global__ __launch_bounds__(256) void cvt_kernel(const float* __restrict__ in,
                                                  __bf16* __restrict__ out, int n) {
    int i = (blockIdx.x * 256 + threadIdx.x) * 4;
    if (i + 3 < n) {
        float4 f = *(const float4*)(in + i);
        out[i + 0] = (__bf16)f.x;
        out[i + 1] = (__bf16)f.y;
        out[i + 2] = (__bf16)f.z;
        out[i + 3] = (__bf16)f.w;
    } else {
        for (int j = i; j < n; ++j) out[j] = (__bf16)in[j];
    }
}

// ---------------------------------------------------------------------------
// Tiled bf16 WMMA GEMM: C[M,N] = A[M,K] * B[K,N] + bias[N]
// MODE 0: QKV epilogue -> scatter bf16 into q[B,H,T,D], k[B,H,T,D], vT[B,H,D,T]
// MODE 1: plain fp32 store to outF
// Block: 256 threads (8 waves), tile BM=128 x BN=64, BK=32 (one WMMA K step)
// ---------------------------------------------------------------------------
template <int MODE>
__global__ __launch_bounds__(256) void gemm_bf16_kernel(
    const __bf16* __restrict__ A,
    const __bf16* __restrict__ Bw,
    const float* __restrict__ bias,
    float* __restrict__ outF,
    __bf16* __restrict__ qg,
    __bf16* __restrict__ kg,
    __bf16* __restrict__ vtg,
    int M, int N, int K)
{
    constexpr int BM = 128, BN = 64, BK = 32;
    __shared__ __bf16 As[BM][BK];   // row-major tile        (8 KB)
    __shared__ __bf16 Bs[BN][BK];   // column-major tile     (4 KB)

    const int tid  = threadIdx.x;
    const int lane = tid & 31;
    const int wave = tid >> 5;          // 0..7
    const int hh   = lane >> 4;         // lane half (0/1)
    const int l15  = lane & 15;

    const int wm = wave >> 1;           // 0..3 -> 32-row strip
    const int wn = wave & 1;            // 0..1 -> 32-col strip
    const int blockM = blockIdx.y * BM;
    const int blockN = blockIdx.x * BN;
    const int waveM  = wm * 32;
    const int waveN  = wn * 32;

    const v8f vzero = {0.f, 0.f, 0.f, 0.f, 0.f, 0.f, 0.f, 0.f};
    v8f acc[2][2];
    #pragma unroll
    for (int i = 0; i < 2; ++i)
        #pragma unroll
        for (int j = 0; j < 2; ++j) acc[i][j] = vzero;

    // cooperative load indices
    const int arow  = tid >> 1;          // 0..127
    const int ahalf = (tid & 1) * 16;    // 0 / 16
    const int bk    = tid >> 3;          // 0..31
    const int bn0   = (tid & 7) * 8;     // 0..56

    for (int k0 = 0; k0 < K; k0 += BK) {
        // ---- stage A tile: 32 B per thread, coalesced ----
        {
            const __bf16* src = A + (size_t)(blockM + arow) * K + k0 + ahalf;
            uint4 v0 = ((const uint4*)src)[0];
            uint4 v1 = ((const uint4*)src)[1];
            *(uint4*)&As[arow][ahalf]     = v0;
            *(uint4*)&As[arow][ahalf + 8] = v1;
            if (k0 + BK < K)
                __builtin_prefetch(src + BK, 0, 0);   // global_prefetch_b8
        }
        // ---- stage B tile (transpose to column-major in LDS) ----
        {
            const __bf16* src = Bw + (size_t)(k0 + bk) * N + blockN + bn0;
            uint4 v = *(const uint4*)src;
            const __bf16* e = (const __bf16*)&v;
            #pragma unroll
            for (int i = 0; i < 8; ++i) Bs[bn0 + i][bk] = e[i];
            if (k0 + BK < K)
                __builtin_prefetch(src + (size_t)BK * N, 0, 0);
        }
        __syncthreads();

        // ---- fragments + 4 WMMAs ----
        FragBF afrag[2], bfrag[2];
        #pragma unroll
        for (int mr = 0; mr < 2; ++mr) {
            int row = waveM + mr * 16 + l15;
            afrag[mr].q[0] = *(const uint4*)&As[row][hh * 8];
            afrag[mr].q[1] = *(const uint4*)&As[row][16 + hh * 8];
        }
        #pragma unroll
        for (int nc = 0; nc < 2; ++nc) {
            int col = waveN + nc * 16 + l15;
            bfrag[nc].q[0] = *(const uint4*)&Bs[col][hh * 8];
            bfrag[nc].q[1] = *(const uint4*)&Bs[col][16 + hh * 8];
        }
        #pragma unroll
        for (int mr = 0; mr < 2; ++mr)
            #pragma unroll
            for (int nc = 0; nc < 2; ++nc)
                acc[mr][nc] = wmma_bf16(afrag[mr].v, bfrag[nc].v, acc[mr][nc]);
        __syncthreads();
    }

    // ---- epilogue ----
    #pragma unroll
    for (int mr = 0; mr < 2; ++mr) {
        #pragma unroll
        for (int nc = 0; nc < 2; ++nc) {
            int gCol = blockN + waveN + nc * 16 + l15;
            float bv = bias[gCol];
            #pragma unroll
            for (int r = 0; r < 8; ++r) {
                int gRow = blockM + waveM + mr * 16 + r + 8 * hh;
                float val = acc[mr][nc][r] + bv;
                if (MODE == 1) {
                    outF[(size_t)gRow * N + gCol] = val;
                } else {
                    int bb = gRow >> 11;            // /T_
                    int tt = gRow & (T_ - 1);
                    int region = gCol >> 10;        // 0:q 1:k 2:v
                    int c = gCol & (C_ - 1);
                    int head = c >> 6;
                    int d = c & (D_ - 1);
                    __bf16 e = (__bf16)val;
                    size_t bhh = (size_t)bb * H_ + head;
                    if (region == 0)
                        qg[(bhh * T_ + tt) * D_ + d] = e;
                    else if (region == 1)
                        kg[(bhh * T_ + tt) * D_ + d] = e;
                    else
                        vtg[(bhh * D_ + d) * T_ + tt] = e;
                }
            }
        }
    }
}

// ---------------------------------------------------------------------------
// Flash-style attention with ALiBi bias (non-causal, matching reference).
// Block: 128 threads = 4 waves; each wave owns 16 query rows.
// K/V chunks are double-buffered in LDS via async global->LDS copies
// (GLOBAL_LOAD_ASYNC_TO_LDS_B128 + s_wait_asynccnt); next chunk is prefetched
// while the current one feeds the WMMAs.
// ---------------------------------------------------------------------------
__global__ __launch_bounds__(128) void attn_kernel(
    const __bf16* __restrict__ qg,
    const __bf16* __restrict__ kg,
    const __bf16* __restrict__ vtg,
    __bf16* __restrict__ yg)
{
    __shared__ __bf16 Ks[2][64][64];   // [buf][s_local][d]       (16 KB)
    __shared__ __bf16 Vs[2][64][64];   // [buf][d][s_local]       (16 KB)
    __shared__ __bf16 Ps[4][16][64];   // per-wave P tile         ( 8 KB)

    const int tid  = threadIdx.x;
    const int lane = tid & 31;
    const int wave = tid >> 5;         // 0..3
    const int hh   = lane >> 4;
    const int l15  = lane & 15;

    const int t0 = blockIdx.x * 64 + wave * 16;
    const int bh = blockIdx.y;         // b*H + h
    const int h  = bh & (H_ - 1);
    const int b  = bh >> 4;

    const float scale = 0.125f;                        // 1/sqrt(64)
    const float slope = exp2f(-0.5f * (float)(h + 1)); // ALiBi slopes for H=16

    const __bf16* qbase = qg  + (size_t)bh * T_ * D_;
    const __bf16* kbase = kg  + (size_t)bh * T_ * D_;
    const __bf16* vbase = vtg + (size_t)bh * D_ * T_;

    // staging indices: K chunk is fully contiguous (rows [s0..s0+63] x D),
    // V chunk is 64 rows of 64 (row stride T_)
    const int vrow  = tid >> 1;           // 0..63 (d index)
    const int vhalf = (tid & 1) * 32;     // 0 / 32

    // Q fragments (row = t0 + l15), K-dim = d split into two 32-chunks
    FragBF qf[2];
    {
        const __bf16* qrow = qbase + (size_t)(t0 + l15) * D_;
        #pragma unroll
        for (int kd = 0; kd < 2; ++kd) {
            qf[kd].q[0] = *(const uint4*)(qrow + kd * 32 + hh * 8);
            qf[kd].q[1] = *(const uint4*)(qrow + kd * 32 + 16 + hh * 8);
        }
    }

    const v8f vzero = {0.f, 0.f, 0.f, 0.f, 0.f, 0.f, 0.f, 0.f};
    v8f o[4];
    #pragma unroll
    for (int dt = 0; dt < 4; ++dt) o[dt] = vzero;
    float mrow[8], lrow[8];
    #pragma unroll
    for (int r = 0; r < 8; ++r) { mrow[r] = -1e30f; lrow[r] = 0.f; }

    constexpr int NCHUNK = T_ / 64;

    // ---- prologue: stage chunk 0 into buffer 0 (8 async b128 per thread) ----
    {
        const __bf16* gk = kbase + tid * 32;
        __bf16* lk = &Ks[0][0][0] + tid * 32;
        #pragma unroll
        for (int i = 0; i < 4; ++i) ASYNC_COPY_B128(gk + i * 8, lk + i * 8);
        const __bf16* gv = vbase + (size_t)vrow * T_ + vhalf;
        __bf16* lv = &Vs[0][0][0] + vrow * 64 + vhalf;
        #pragma unroll
        for (int i = 0; i < 4; ++i) ASYNC_COPY_B128(gv + i * 8, lv + i * 8);
    }

    for (int ch = 0; ch < NCHUNK; ++ch) {
        const int cur = ch & 1;
        const int s0  = ch * 64;

        // ---- stage next chunk into the other buffer, retire current ----
        if (ch + 1 < NCHUNK) {
            const int nxt = cur ^ 1;
            const int s1  = s0 + 64;
            const __bf16* gk = kbase + s1 * 64 + tid * 32;
            __bf16* lk = &Ks[nxt][0][0] + tid * 32;
            #pragma unroll
            for (int i = 0; i < 4; ++i) ASYNC_COPY_B128(gk + i * 8, lk + i * 8);
            const __bf16* gv = vbase + (size_t)vrow * T_ + s1 + vhalf;
            __bf16* lv = &Vs[nxt][0][0] + vrow * 64 + vhalf;
            #pragma unroll
            for (int i = 0; i < 4; ++i) ASYNC_COPY_B128(gv + i * 8, lv + i * 8);
            ASYNC_WAIT(8);   // current buffer complete; next 8 stay in flight
        } else {
            ASYNC_WAIT(0);
        }
        __syncthreads();

        // ---- scores S = q @ k^T for this wave's 16x64 tile (from LDS) ----
        v8f sacc[4];
        #pragma unroll
        for (int nc = 0; nc < 4; ++nc) sacc[nc] = vzero;
        #pragma unroll
        for (int nc = 0; nc < 4; ++nc) {
            const __bf16* kcol = &Ks[cur][nc * 16 + l15][0];
            #pragma unroll
            for (int kd = 0; kd < 2; ++kd) {
                FragBF kf;
                kf.q[0] = *(const uint4*)(kcol + kd * 32 + hh * 8);
                kf.q[1] = *(const uint4*)(kcol + kd * 32 + 16 + hh * 8);
                sacc[nc] = wmma_bf16(qf[kd].v, kf.v, sacc[nc]);
            }
        }

        // ---- logits with ALiBi, online softmax update ----
        float lg[4][8];
        #pragma unroll
        for (int nc = 0; nc < 4; ++nc) {
            int sg = s0 + nc * 16 + l15;
            #pragma unroll
            for (int r = 0; r < 8; ++r) {
                int tg = t0 + r + 8 * hh;
                int dd = sg - tg;
                float ab = (dd < 0) ? slope * (float)dd : 0.0f;
                lg[nc][r] = sacc[nc][r] * scale + ab;
            }
        }
        #pragma unroll
        for (int r = 0; r < 8; ++r) {
            float mx = fmaxf(fmaxf(lg[0][r], lg[1][r]), fmaxf(lg[2][r], lg[3][r]));
            #pragma unroll
            for (int off = 1; off < 16; off <<= 1)
                mx = fmaxf(mx, __shfl_xor(mx, off, 16));
            float mnew  = fmaxf(mrow[r], mx);
            float alpha = __expf(mrow[r] - mnew);
            mrow[r] = mnew;
            float ssum = 0.f;
            #pragma unroll
            for (int nc = 0; nc < 4; ++nc) {
                float p = __expf(lg[nc][r] - mnew);
                lg[nc][r] = p;
                ssum += p;
            }
            #pragma unroll
            for (int off = 1; off < 16; off <<= 1)
                ssum += __shfl_xor(ssum, off, 16);
            lrow[r] = lrow[r] * alpha + ssum;
            #pragma unroll
            for (int dt = 0; dt < 4; ++dt) o[dt][r] *= alpha;
        }

        // ---- spill P to LDS (C layout -> A layout re-read) ----
        #pragma unroll
        for (int nc = 0; nc < 4; ++nc)
            #pragma unroll
            for (int r = 0; r < 8; ++r)
                Ps[wave][r + 8 * hh][nc * 16 + l15] = (__bf16)lg[nc][r];
        __syncthreads();

        // ---- O += P @ V_chunk (from LDS) ----
        FragBF pf[2];
        #pragma unroll
        for (int kd = 0; kd < 2; ++kd) {
            pf[kd].q[0] = *(const uint4*)&Ps[wave][l15][kd * 32 + hh * 8];
            pf[kd].q[1] = *(const uint4*)&Ps[wave][l15][kd * 32 + 16 + hh * 8];
        }
        #pragma unroll
        for (int dt = 0; dt < 4; ++dt) {
            const __bf16* vcol = &Vs[cur][dt * 16 + l15][0];
            #pragma unroll
            for (int kd = 0; kd < 2; ++kd) {
                FragBF vf;
                vf.q[0] = *(const uint4*)(vcol + kd * 32 + hh * 8);
                vf.q[1] = *(const uint4*)(vcol + kd * 32 + 16 + hh * 8);
                o[dt] = wmma_bf16(pf[kd].v, vf.v, o[dt]);
            }
        }
        __syncthreads();   // protect LDS buffers before next staging
    }

    // ---- normalize + store y as bf16 [B,T,C] for projection GEMM ----
    #pragma unroll
    for (int r = 0; r < 8; ++r) {
        float inv = 1.0f / lrow[r];
        int tg = t0 + r + 8 * hh;
        __bf16* ydst = yg + ((size_t)b * T_ + tg) * C_ + h * D_;
        #pragma unroll
        for (int dt = 0; dt < 4; ++dt)
            ydst[dt * 16 + l15] = (__bf16)(o[dt][r] * inv);
    }
}

// ---------------------------------------------------------------------------
// Host-side launcher
// ---------------------------------------------------------------------------
extern "C" void kernel_launch(void* const* d_in, const int* in_sizes, int n_in,
                              void* d_out, int out_size, void* d_ws, size_t ws_size,
                              hipStream_t stream) {
    const float* x      = (const float*)d_in[0];
    const float* W_qkv  = (const float*)d_in[1];
    const float* b_qkv  = (const float*)d_in[2];
    const float* W_proj = (const float*)d_in[3];
    const float* b_proj = (const float*)d_in[4];
    float* out = (float*)d_out;

    const size_t nX     = (size_t)B_ * T_ * C_;       // 4,194,304
    const size_t nWqkv  = (size_t)C_ * 3 * C_;        // 3,145,728
    const size_t nWproj = (size_t)C_ * C_;            // 1,048,576
    const size_t nHead  = (size_t)B_ * H_ * T_ * D_;  // 4,194,304

    __bf16* xb  = (__bf16*)d_ws;
    __bf16* wqb = xb + nX;
    __bf16* wpb = wqb + nWqkv;
    __bf16* qb  = wpb + nWproj;
    __bf16* kb  = qb + nHead;
    __bf16* vtb = kb + nHead;
    __bf16* yb  = vtb + nHead;

    cvt_kernel<<<dim3((unsigned)(nX / 1024)), 256, 0, stream>>>(x, xb, (int)nX);
    cvt_kernel<<<dim3((unsigned)(nWqkv / 1024)), 256, 0, stream>>>(W_qkv, wqb, (int)nWqkv);
    cvt_kernel<<<dim3((unsigned)(nWproj / 1024)), 256, 0, stream>>>(W_proj, wpb, (int)nWproj);

    // QKV GEMM: M=4096, N=3072, K=1024
    gemm_bf16_kernel<0><<<dim3(3 * C_ / 64, (B_ * T_) / 128), 256, 0, stream>>>(
        xb, wqb, b_qkv, nullptr, qb, kb, vtb, B_ * T_, 3 * C_, C_);

    // Attention: grid (T/64, B*H)
    attn_kernel<<<dim3(T_ / 64, B_ * H_), 128, 0, stream>>>(qb, kb, vtb, yb);

    // Projection GEMM: M=4096, N=1024, K=1024, fp32 output
    gemm_bf16_kernel<1><<<dim3(C_ / 64, (B_ * T_) / 128), 256, 0, stream>>>(
        yb, wpb, b_proj, out, nullptr, nullptr, nullptr, B_ * T_, C_, C_);
}
